// LSTMModel_69724499083934
// MI455X (gfx1250) — matrix operand order repaired and checked by reading, not compiled, not run on hardware
//
#include <hip/hip_runtime.h>
#include <hip/hip_bf16.h>

typedef __attribute__((ext_vector_type(16))) __bf16 v16bf;
typedef __attribute__((ext_vector_type(8)))  __bf16 v8bf;
typedef __attribute__((ext_vector_type(8)))  float  v8f;

#define B_   512
#define T_   512
#define F_   64
#define H1_  32
#define G1_  128   // 4*H1
#define H2_  16
#define G2_  64    // 4*H2
#define OUT_ 12
#define NTHREADS 256

__device__ __forceinline__ float sigmoid_(float x) {
  return __builtin_amdgcn_rcpf(1.0f + __expf(-x));
}
// branch-free tanh: copysign((1-e)/(1+e), x), e = exp(-2|x|)
__device__ __forceinline__ float tanh_fast(float x) {
  float e = __expf(-2.0f * __builtin_fabsf(x));
  float t = (1.0f - e) * __builtin_amdgcn_rcpf(1.0f + e);
  return __builtin_copysignf(t, x);
}

// A operand (16x32 bf16) from row-major LDS [16][stride], K-window at kblock.
// ISA 7.12.2: lanes 0-15 hold row M=lane, K={0..7}+{16..23}; lanes 16-31 K={8..15}+{24..31}.
__device__ __forceinline__ v16bf load_A(const __bf16* base, int stride, int kblock, int lane) {
  int row  = lane & 15;
  int koff = kblock + ((lane >> 4) << 3);
  const __bf16* p = base + row * stride + koff;
  v8bf lo = *(const v8bf*)(p);        // one ds_load_b128
  v8bf hi = *(const v8bf*)(p + 16);
  v16bf a;
  #pragma unroll
  for (int j = 0; j < 8; ++j) { a[j] = lo[j]; a[8 + j] = hi[j]; }
  return a;
}

// B operand from pre-swizzled LDS: [op][lane][16] contiguous -> 2x ds_load_b128,
// values land packed in contiguous VGPR quads (no repack movs).
__device__ __forceinline__ v16bf load_Bp(const __bf16* opbase, int lane) {
  const __bf16* p = opbase + lane * 16;
  v8bf lo = *(const v8bf*)(p);
  v8bf hi = *(const v8bf*)(p + 8);
  v16bf b;
  #pragma unroll
  for (int j = 0; j < 8; ++j) { b[j] = lo[j]; b[8 + j] = hi[j]; }
  return b;
}

__global__ __launch_bounds__(NTHREADS) void lstm_fused(
    const float* __restrict__ x,
    const float* __restrict__ W1, const float* __restrict__ U1, const float* __restrict__ b1,
    const float* __restrict__ W2, const float* __restrict__ U2, const float* __restrict__ b2,
    const float* __restrict__ Wd, const float* __restrict__ bd,
    float* __restrict__ out)
{
  // weights pre-swizzled into WMMA-B operand layout: [op][lane][j]
  __shared__ __align__(16) __bf16 w1b[2 * 8 * 32 * 16]; // 16 KB  (kb, nt)
  __shared__ __align__(16) __bf16 u1b[8 * 32 * 16];     // 8 KB
  __shared__ __align__(16) __bf16 w2b[4 * 32 * 16];     // 4 KB
  __shared__ __align__(16) __bf16 u2b[4 * 32 * 16];     // 4 KB (K rows 16..31 = 0)
  __shared__ __align__(16) __bf16 xsb[2][16 * F_];      // 2 x 2 KB, double-buffered x_t
  __shared__ __align__(16) __bf16 h1s[16 * H1_];        // 1 KB
  __shared__ __align__(16) __bf16 h2s[16 * H1_];        // 1 KB (cols 16..31 = 0)
  // WMMA-native z layout: z[ntile][lane][r]  (lane stores 8 contiguous f32)
  __shared__ __align__(32) float z1w[8 * 32 * 8];       // 8 KB
  __shared__ __align__(32) float z2w[4 * 32 * 8];       // 4 KB
  __shared__ float h2f[16 * H2_];

  const int tid  = threadIdx.x;
  const int lane = tid & 31;
  const int wave = tid >> 5;
  const int b0   = blockIdx.x * 16;
  const size_t xrow = (size_t)T_ * F_;

  // ---- one-time: stage weights into LDS pre-swizzled to B layout ----
  // W1: K=64 (2 kblocks), N=128 (8 ntiles)
  for (int idx = tid; idx < 2 * 8 * 32 * 16; idx += NTHREADS) {
    int j = idx & 15, l = (idx >> 4) & 31, nt = (idx >> 9) & 7, kb = idx >> 12;
    int krow = kb * 32 + ((l >> 4) << 3) + (j & 7) + ((j & 8) << 1);
    int col  = nt * 16 + (l & 15);
    w1b[idx] = (__bf16)W1[krow * G1_ + col];
  }
  // U1: K=32, N=128
  for (int idx = tid; idx < 8 * 32 * 16; idx += NTHREADS) {
    int j = idx & 15, l = (idx >> 4) & 31, nt = idx >> 9;
    int krow = ((l >> 4) << 3) + (j & 7) + ((j & 8) << 1);
    int col  = nt * 16 + (l & 15);
    u1b[idx] = (__bf16)U1[krow * G1_ + col];
  }
  // W2: K=32, N=64
  for (int idx = tid; idx < 4 * 32 * 16; idx += NTHREADS) {
    int j = idx & 15, l = (idx >> 4) & 31, nt = idx >> 9;
    int krow = ((l >> 4) << 3) + (j & 7) + ((j & 8) << 1);
    int col  = nt * 16 + (l & 15);
    w2b[idx] = (__bf16)W2[krow * G2_ + col];
  }
  // U2: K=16 zero-padded to 32
  for (int idx = tid; idx < 4 * 32 * 16; idx += NTHREADS) {
    int j = idx & 15, l = (idx >> 4) & 31, nt = idx >> 9;
    int krow = ((l >> 4) << 3) + (j & 7) + ((j & 8) << 1);
    int col  = nt * 16 + (l & 15);
    u2b[idx] = (krow < H2_) ? (__bf16)U2[krow * G2_ + col] : (__bf16)0.0f;
  }
  for (int i = tid; i < 16 * H1_; i += NTHREADS) { h1s[i] = (__bf16)0.0f; h2s[i] = (__bf16)0.0f; }
  // stage x(t=0): 256 thr x 4 elems
  {
    int e = tid * 4;
    int r = e >> 6, c = e & 63;
    const float4 v = *(const float4*)(x + (size_t)(b0 + r) * xrow + c);
    xsb[0][e + 0] = (__bf16)v.x; xsb[0][e + 1] = (__bf16)v.y;
    xsb[0][e + 2] = (__bf16)v.z; xsb[0][e + 3] = (__bf16)v.w;
  }
  __syncthreads();

  // ---- hoist loop-invariant B-operands (weights) and bias accumulators ----
  v16bf Bw1a = load_Bp(w1b + (0 * 8 + wave) * 512, lane);
  v16bf Bw1b = load_Bp(w1b + (1 * 8 + wave) * 512, lane);
  v16bf Bu1  = load_Bp(u1b + wave * 512, lane);
  v16bf Bw2 = {}, Bu2 = {};
  float accb2 = 0.0f;
  if (wave < 4) {
    Bw2 = load_Bp(w2b + wave * 512, lane);
    Bu2 = load_Bp(u2b + wave * 512, lane);
    accb2 = b2[wave * 16 + (lane & 15)];
  }
  const float accb1 = b1[wave * 16 + (lane & 15)];
  // fully-splatted C-operand init vectors; pin them in VGPRs so LLVM cannot
  // rematerialize the bias load + splat inside the t-loop
  v8f accInit1 = {accb1, accb1, accb1, accb1, accb1, accb1, accb1, accb1};
  v8f accInit2 = {accb2, accb2, accb2, accb2, accb2, accb2, accb2, accb2};
  asm volatile("" : "+v"(accInit1), "+v"(accInit2));
  const v8f accZero = {0.f, 0.f, 0.f, 0.f, 0.f, 0.f, 0.f, 0.f};

  // per-thread persistent cell state (fixed item ownership across all timesteps)
  float c1r0 = 0.0f, c1r1 = 0.0f, c2r = 0.0f;

  for (int t = 0; t < T_; ++t) {
    const __bf16* xcur = xsb[t & 1];

    // ---- region 1: layer-2 gates for t-1 (disjoint buffers) + z1 WMMA for t ----
    if (t > 0) {
      int rr = tid >> 4, hh = tid & 15;           // 16x16 items, 1/thread
      int bidx = (hh | ((rr & 8) << 1)) * 8 + (rr & 7);
      float iv = sigmoid_(z2w[0 * 256 + bidx]);
      float fv = sigmoid_(z2w[1 * 256 + bidx]);
      float gv = fmaxf  (z2w[2 * 256 + bidx], 0.0f);
      float ov = sigmoid_(z2w[3 * 256 + bidx]);
      float c  = fv * c2r + iv * gv;
      c2r = c;
      float h  = ov * fmaxf(c, 0.0f);
      h2s[rr * H1_ + hh] = (__bf16)h;
      h2f[tid] = h;
    }
    {
      // two independent accumulator chains; bias enters via pinned C operand
      v16bf a0 = load_A(xcur, F_, 0, lane);
      v16bf a1 = load_A(xcur, F_, 32, lane);
      v16bf ah = load_A(h1s, H1_, 0, lane);
      v8f accA = __builtin_amdgcn_wmma_f32_16x16x32_bf16(false, a0, false, Bw1a, (short)0, accInit1, false, false);
      v8f accB = __builtin_amdgcn_wmma_f32_16x16x32_bf16(false, a1, false, Bw1b, (short)0, accZero,  false, false);
      accA     = __builtin_amdgcn_wmma_f32_16x16x32_bf16(false, ah, false, Bu1,  (short)0, accA,     false, false);
      #pragma unroll
      for (int r = 0; r < 8; ++r) accA[r] += accB[r];
      *(v8f*)(z1w + wave * 256 + lane * 8) = accA; // 2x ds_store_b128
    }
    __syncthreads();

    // ---- region 2: layer-1 gates (i,f,g,o Keras order), activation=tanh ----
    #pragma unroll
    for (int k = 0; k < 2; ++k) {
      int i  = tid + k * NTHREADS;                // 16x32 items, 2/thread
      int rr = i >> 5, hh = i & 31;
      int bidx = ((hh & 15) | ((rr & 8) << 1)) * 8 + (rr & 7);
      int nt = hh >> 4;
      float iv = sigmoid_(z1w[(nt + 0) * 256 + bidx]);
      float fv = sigmoid_(z1w[(nt + 2) * 256 + bidx]);
      float gv = tanh_fast(z1w[(nt + 4) * 256 + bidx]);
      float ov = sigmoid_(z1w[(nt + 6) * 256 + bidx]);
      float cprev = k ? c1r1 : c1r0;
      float c = fv * cprev + iv * gv;
      if (k) c1r1 = c; else c1r0 = c;
      h1s[i] = (__bf16)(ov * tanh_fast(c));
    }
    __syncthreads();

    // ---- region 3: z2 WMMA (waves 0-3)  ||  stage x(t+1) (waves 4-7) ----
    if (wave < 4) {
      v16bf a0 = load_A(h1s, H1_, 0, lane);
      v16bf a1 = load_A(h2s, H1_, 0, lane);
      v8f accA = __builtin_amdgcn_wmma_f32_16x16x32_bf16(false, a0, false, Bw2, (short)0, accInit2, false, false);
      v8f accB = __builtin_amdgcn_wmma_f32_16x16x32_bf16(false, a1, false, Bu2, (short)0, accZero,  false, false);
      #pragma unroll
      for (int r = 0; r < 8; ++r) accA[r] += accB[r];
      *(v8f*)(z2w + wave * 256 + lane * 8) = accA;
    } else if (t + 1 < T_) {
      int e = (tid - 128) * 8;                    // 128 thr x 8 elems = 1024
      int r = e >> 6, c = e & 63;
      const float* px = x + (size_t)(b0 + r) * xrow + (size_t)(t + 1) * F_ + c;
      float4 v0 = *(const float4*)(px);
      float4 v1 = *(const float4*)(px + 4);
      __bf16* q = &xsb[(t + 1) & 1][e];
      q[0] = (__bf16)v0.x; q[1] = (__bf16)v0.y; q[2] = (__bf16)v0.z; q[3] = (__bf16)v0.w;
      q[4] = (__bf16)v1.x; q[5] = (__bf16)v1.y; q[6] = (__bf16)v1.z; q[7] = (__bf16)v1.w;
    }
    __syncthreads();
  }

  // ---- final layer-2 gates (t = T-1) ----
  {
    int rr = tid >> 4, hh = tid & 15;
    int bidx = (hh | ((rr & 8) << 1)) * 8 + (rr & 7);
    float iv = sigmoid_(z2w[0 * 256 + bidx]);
    float fv = sigmoid_(z2w[1 * 256 + bidx]);
    float gv = fmaxf  (z2w[2 * 256 + bidx], 0.0f);
    float ov = sigmoid_(z2w[3 * 256 + bidx]);
    float c  = fv * c2r + iv * gv;
    h2f[tid] = ov * fmaxf(c, 0.0f);
  }
  __syncthreads();

  // ---- dense head: out[b, :12] = h2_last @ Wd + bd ----
  if (tid < 16 * OUT_) {
    int r = tid / OUT_, o = tid % OUT_;
    float s = bd[o];
    #pragma unroll
    for (int h = 0; h < H2_; ++h) s += h2f[r * H2_ + h] * Wd[h * OUT_ + o];
    out[(size_t)(b0 + r) * OUT_ + o] = s;
  }
}

extern "C" void kernel_launch(void* const* d_in, const int* in_sizes, int n_in,
                              void* d_out, int out_size, void* d_ws, size_t ws_size,
                              hipStream_t stream) {
  const float* x  = (const float*)d_in[0];
  const float* W1 = (const float*)d_in[1];
  const float* U1 = (const float*)d_in[2];
  const float* b1 = (const float*)d_in[3];
  const float* W2 = (const float*)d_in[4];
  const float* U2 = (const float*)d_in[5];
  const float* b2 = (const float*)d_in[6];
  const float* Wd = (const float*)d_in[7];
  const float* bd = (const float*)d_in[8];
  float* out = (float*)d_out;
  (void)in_sizes; (void)n_in; (void)out_size; (void)d_ws; (void)ws_size;
  dim3 grid(B_ / 16), block(NTHREADS);
  hipLaunchKernelGGL(lstm_fused, grid, block, 0, stream,
                     x, W1, U1, b1, W2, U2, b2, Wd, bd, out);
}
